// PyTorchMSDeformAttn_24507083391017
// MI455X (gfx1250) — matrix operand order repaired and checked by reading, not compiled, not run on hardware
//
#include <hip/hip_runtime.h>
#include <hip/hip_bf16.h>
#include <stdint.h>

// ---------------------------------------------------------------------------
// MSDeformAttn forward for MI455X (gfx1250, wave32)
//   - WMMA f32_16x16x32_f16 GEMMs
//   - B operand staged in LDS via async global->LDS copies (ASYNCcnt path)
// ---------------------------------------------------------------------------

typedef __attribute__((ext_vector_type(16))) _Float16 v16h;
typedef __attribute__((ext_vector_type(8)))  float    v8f;

#define D_MODEL   256
#define N_HEADS   8
#define N_LEVELS  4
#define N_POINTS  4
#define HEAD_DIM  32
#define LEN_IN    5440
#define LEN_Q     5440
#define BATCH     4
#define M_ROWS    (BATCH * LEN_Q)   // 21760 rows for every GEMM
#define K_DIM     256
#define K_STEPS   (K_DIM / 32)      // 8

// ------------------------------- converts ---------------------------------

__global__ void cvt_f32_f16(const float* __restrict__ src,
                            _Float16* __restrict__ dst, int n) {
  for (int i = blockIdx.x * blockDim.x + threadIdx.x; i < n;
       i += gridDim.x * blockDim.x)
    dst[i] = (_Float16)src[i];
}

// Wt[n][k] = (f16) W[k][n]   (W is K x N row-major fp32)
__global__ void transpose_cvt(const float* __restrict__ W,
                              _Float16* __restrict__ Wt, int K, int N) {
  int i = blockIdx.x * blockDim.x + threadIdx.x;
  if (i < N * K) {
    int n = i / K, k = i % K;
    Wt[i] = (_Float16)W[(size_t)k * N + n];
  }
}

// ------------------------------ WMMA GEMM ----------------------------------
// C[M][N] = A[M][256] * W[256][N] + bias[N]
// A: f16 row-major (lda=256).  Wt: f16 N x 256 row-major (transposed weight).
// Grid: (M/128, N/64). Block = 8 waves (256 threads); wave w computes the
// 16-row tile m = blockIdx.x*128 + w*16 over the block's 64-column N slice.
// The 32KB contiguous Wt slice for this block is staged into LDS with async
// global->LDS b128 copies (ASYNCcnt), then every B fragment is a 32B ds read.
// A is streamed from global, double-buffered in registers.
template <int N>
__global__ __launch_bounds__(256)
void gemm_wmma(const _Float16* __restrict__ A, const _Float16* __restrict__ Wt,
               const float* __restrict__ bias, float* __restrict__ C) {
  __shared__ _Float16 smem[64 * K_DIM];  // 32 KB: Wt rows [nbase, nbase+64)

  const int tid  = threadIdx.x;
  const int wave = tid >> 5;
  const int lane = tid & 31;
  const int mbase = blockIdx.x * 128 + wave * 16;
  const int nbase = blockIdx.y * 64;
  const int kg = lane >> 4;   // lane group: selects 16-wide K half (contiguous k = 16*kg + s)
  const int mr = lane & 15;   // A row within tile / B,C column within tile

  // ---- async stage of the block's Wt slice into LDS ----------------------
  {
    typedef _Float16 __attribute__((address_space(3))) lds_h;
    uint32_t lbase = (uint32_t)(uintptr_t)(lds_h*)smem;
    uint64_t gbase = (uint64_t)(uintptr_t)(Wt + (size_t)nbase * K_DIM);
#pragma unroll
    for (int j = 0; j < 8; ++j) {
      uint32_t lo = lbase + (uint32_t)((j * 256 + tid) * 16);
      uint64_t go = gbase + (uint64_t)((j * 256 + tid) * 16);
      asm volatile("global_load_async_to_lds_b128 %0, %1, off"
                   :: "v"(lo), "v"(go) : "memory");
    }
    asm volatile("s_wait_asynccnt 0x0" ::: "memory");
  }
  __syncthreads();

  // ---- K loop: A double-buffered from global, B fragments from LDS -------
  const _Float16* __restrict__ arow = A + (size_t)(mbase + mr) * K_DIM + kg * 16;
  const _Float16* __restrict__ bsm  = smem + (size_t)mr * K_DIM + kg * 16;

  union AF { v16h v; uint4 u[2]; };
  AF  a[2];
  v8f acc[4] = {v8f{}, v8f{}, v8f{}, v8f{}};

  a[0].u[0] = *(const uint4*)(arow + 0);
  a[0].u[1] = *(const uint4*)(arow + 8);

#pragma unroll
  for (int s = 0; s < K_STEPS; ++s) {
    const int cur = s & 1, nxt = cur ^ 1;
    if (s + 1 < K_STEPS) {
      const int k1 = (s + 1) * 32;
      __builtin_prefetch(arow + k1 + 32, 0, 3);   // global_prefetch_b8
      a[nxt].u[0] = *(const uint4*)(arow + k1);
      a[nxt].u[1] = *(const uint4*)(arow + k1 + 8);
    }
#pragma unroll
    for (int f = 0; f < 4; ++f) {
      // B fragment: 16 contiguous halves from the LDS-resident Wt slice.
      v16h b = *(const v16h*)(bsm + (size_t)f * 16 * K_DIM + s * 32);
      acc[f] = __builtin_amdgcn_wmma_f32_16x16x32_f16(
          /*neg_a=*/false, a[cur].v, /*neg_b=*/false, b,
          /*c_mod=*/(short)0, acc[f], /*reuse_a=*/false, /*reuse_b=*/false);
    }
  }

#pragma unroll
  for (int f = 0; f < 4; ++f) {
    const int n = nbase + f * 16 + mr;
    const float bs = bias[n];
    float* cp = C + (size_t)(mbase + 8 * kg) * N + n;
#pragma unroll
    for (int r = 0; r < 8; ++r)
      cp[(size_t)r * N] = acc[f][r] + bs;
  }
}

// --------------------------- deformable sampling ---------------------------
// One wave per (b, q, head); lane = channel d (0..31). Control flow is
// wave-uniform (all lanes share the same sample point), corner gathers are
// 32-lane coalesced 128B reads from value[b][idx][h*32 + d].
__global__ __launch_bounds__(256)
void msda_sample(const float* __restrict__ value,      // [B][LEN_IN][256] f32
                 const float* __restrict__ attn_logit, // [B][Lq][128] f32
                 const float* __restrict__ off,        // [B][Lq][256] f32
                 const float* __restrict__ refp,       // [B][Lq][4][2] f32
                 _Float16* __restrict__ midh) {        // [B][Lq][256] f16
  const int tid  = threadIdx.x;
  const int lane = tid & 31;
  const int wid  = blockIdx.x * 8 + (tid >> 5);  // global wave id
  const int h  = wid & 7;
  const int bq = wid >> 3;                       // b * Lq + q
  const int b  = bq / LEN_Q;

  // softmax over the 16 (level,point) logits for this head
  const float* lg = attn_logit + (size_t)bq * (N_HEADS * 16) + h * 16;
  float w[16];
  float mx = -3.0e38f;
#pragma unroll
  for (int j = 0; j < 16; ++j) { w[j] = lg[j]; mx = fmaxf(mx, w[j]); }
  float s = 0.f;
#pragma unroll
  for (int j = 0; j < 16; ++j) { w[j] = __expf(w[j] - mx); s += w[j]; }
  const float inv = 1.0f / s;

  const float* op = off  + (size_t)bq * (N_HEADS * 32) + h * 32;
  const float* rp = refp + (size_t)bq * (N_LEVELS * 2);

  constexpr int HL[4]    = {64, 32, 16, 8};
  constexpr int WL[4]    = {64, 32, 16, 8};
  constexpr int START[4] = {0, 4096, 5120, 5376};

  float acc = 0.f;
#pragma unroll
  for (int lvl = 0; lvl < N_LEVELS; ++lvl) {
    const int   iw = WL[lvl], ih = HL[lvl];
    const float fw = (float)iw, fh = (float)ih;
    const float rx = rp[lvl * 2 + 0], ry = rp[lvl * 2 + 1];
    const float* __restrict__ vb =
        value + ((size_t)b * LEN_IN + START[lvl]) * D_MODEL + h * HEAD_DIM + lane;
#pragma unroll
    for (int p = 0; p < N_POINTS; ++p) {
      const float ox = op[(lvl * 4 + p) * 2 + 0];
      const float oy = op[(lvl * 4 + p) * 2 + 1];
      // grid = 2*(ref + off/norm) - 1 ; px = (gx+1)*W/2 - 0.5 = rx*W + ox - 0.5
      const float px = rx * fw + ox - 0.5f;
      const float py = ry * fh + oy - 0.5f;
      const float aw = w[lvl * 4 + p] * inv;

      const float x0f = floorf(px), y0f = floorf(py);
      const int   x0 = (int)x0f, y0 = (int)y0f;
      const float wx = px - x0f, wy = py - y0f;

      float v00 = 0.f, v10 = 0.f, v01 = 0.f, v11 = 0.f;
      const bool xin0 = (x0 >= 0) & (x0 < iw);
      const bool xin1 = (x0 + 1 >= 0) & (x0 + 1 < iw);
      const bool yin0 = (y0 >= 0) & (y0 < ih);
      const bool yin1 = (y0 + 1 >= 0) & (y0 + 1 < ih);
      if (yin0 & xin0) v00 = vb[(size_t)(y0 * iw + x0) * D_MODEL];
      if (yin0 & xin1) v10 = vb[(size_t)(y0 * iw + x0 + 1) * D_MODEL];
      if (yin1 & xin0) v01 = vb[(size_t)((y0 + 1) * iw + x0) * D_MODEL];
      if (yin1 & xin1) v11 = vb[(size_t)((y0 + 1) * iw + x0 + 1) * D_MODEL];

      acc += aw * ((1.f - wx) * (1.f - wy) * v00 + wx * (1.f - wy) * v10 +
                   (1.f - wx) * wy * v01 + wx * wy * v11);
    }
  }
  midh[(size_t)bq * D_MODEL + h * HEAD_DIM + lane] = (_Float16)acc;
}

// ------------------------------- launcher ----------------------------------

static inline char* align256(char* p) {
  return (char*)(((uintptr_t)p + 255) & ~(uintptr_t)255);
}

extern "C" void kernel_launch(void* const* d_in, const int* in_sizes, int n_in,
                              void* d_out, int out_size, void* d_ws,
                              size_t ws_size, hipStream_t stream) {
  (void)in_sizes; (void)n_in; (void)out_size; (void)ws_size;

  const float* query   = (const float*)d_in[0];
  const float* refp    = (const float*)d_in[1];
  const float* in_flat = (const float*)d_in[2];
  // d_in[3], d_in[4]: spatial shapes / level starts (static, hardcoded)
  const float* W_val  = (const float*)d_in[5];
  const float* b_val  = (const float*)d_in[6];
  const float* W_off  = (const float*)d_in[7];
  const float* b_off  = (const float*)d_in[8];
  const float* W_attn = (const float*)d_in[9];
  const float* b_attn = (const float*)d_in[10];
  const float* W_out  = (const float*)d_in[11];
  const float* b_out  = (const float*)d_in[12];
  float* out = (float*)d_out;

  // workspace layout
  char* p = (char*)d_ws;
  _Float16* q_h  = (_Float16*)align256(p); p = (char*)(q_h  + (size_t)M_ROWS * K_DIM);
  _Float16* in_h = (_Float16*)align256(p); p = (char*)(in_h + (size_t)M_ROWS * K_DIM);
  _Float16* Wv_t = (_Float16*)align256(p); p = (char*)(Wv_t + 256 * 256);
  _Float16* Wo_t = (_Float16*)align256(p); p = (char*)(Wo_t + 256 * 256);
  _Float16* Wa_t = (_Float16*)align256(p); p = (char*)(Wa_t + 128 * 256);
  _Float16* Wu_t = (_Float16*)align256(p); p = (char*)(Wu_t + 256 * 256);
  float* value = (float*)align256(p); p = (char*)(value + (size_t)M_ROWS * 256);
  float* offb  = (float*)align256(p); p = (char*)(offb  + (size_t)M_ROWS * 256);
  float* attn  = (float*)align256(p); p = (char*)(attn  + (size_t)M_ROWS * 128);
  _Float16* midh = (_Float16*)align256(p);

  const int NEL = M_ROWS * K_DIM;  // 5,570,560

  // 1) f32 -> f16 conversions
  cvt_f32_f16<<<(NEL + 255) / 256, 256, 0, stream>>>(query,   q_h,  NEL);
  cvt_f32_f16<<<(NEL + 255) / 256, 256, 0, stream>>>(in_flat, in_h, NEL);
  transpose_cvt<<<(256 * 256 + 255) / 256, 256, 0, stream>>>(W_val,  Wv_t, 256, 256);
  transpose_cvt<<<(256 * 256 + 255) / 256, 256, 0, stream>>>(W_off,  Wo_t, 256, 256);
  transpose_cvt<<<(128 * 256 + 255) / 256, 256, 0, stream>>>(W_attn, Wa_t, 256, 128);
  transpose_cvt<<<(256 * 256 + 255) / 256, 256, 0, stream>>>(W_out,  Wu_t, 256, 256);

  // 2) projection GEMMs (WMMA, B staged in LDS via async copies)
  gemm_wmma<256><<<dim3(M_ROWS / 128, 4), 256, 0, stream>>>(in_h, Wv_t, b_val,  value);
  gemm_wmma<256><<<dim3(M_ROWS / 128, 4), 256, 0, stream>>>(q_h,  Wo_t, b_off,  offb);
  gemm_wmma<128><<<dim3(M_ROWS / 128, 2), 256, 0, stream>>>(q_h,  Wa_t, b_attn, attn);

  // 3) softmax + deformable bilinear sampling: one wave per (b,q,head)
  msda_sample<<<(BATCH * LEN_Q * N_HEADS) / 8, 256, 0, stream>>>(
      value, attn, offb, refp, midh);

  // 4) output projection into d_out
  gemm_wmma<256><<<dim3(M_ROWS / 128, 4), 256, 0, stream>>>(midh, Wu_t, b_out, out);
}
